// SingleESN_90486370992874
// MI455X (gfx1250) — matrix operand order repaired and checked by reading, not compiled, not run on hardware
//
#include <hip/hip_runtime.h>
#include <hip/hip_bf16.h>
#include <math.h>

// ---------------------------------------------------------------------------
// ESN: u_proj = x @ Win  (bf16 WMMA GEMM; A tile staged by the Tensor Data
//      Mover, B tile converted+transposed by the waves)
//      v_t   = (1-a)*v + a*tanh(v @ W + u_t)  (persistent kernel, grid sync)
// ---------------------------------------------------------------------------

#define ALPHA   0.5f
#define T_STEPS 4096
#define NIN     8192
#define NOUT    2048

typedef __attribute__((ext_vector_type(16))) __bf16       v16bf;
typedef __attribute__((ext_vector_type(8)))  float        v8f;
typedef __attribute__((ext_vector_type(4)))  unsigned int v4u;
typedef __attribute__((ext_vector_type(8)))  int          v8i;
typedef __attribute__((ext_vector_type(4)))  int          v4i;

// ---------------------------------------------------------------------------
// Phase 1: bf16 WMMA GEMM.  Block tile 128(M) x 64(N), K-step 32.
// 8 waves per block, 4x2 wave grid, each wave owns a 32x32 region (2x2 WMMA).
// LDS: A as raw f32 [M=128][K=32] row-major, DMA'd in by the TDM;
//      B transposed bf16 [N=64][K=32] (B frag = one contiguous 32B chunk).
// ---------------------------------------------------------------------------
__global__ __launch_bounds__(256)
void esn_gemm_bf16(const float* __restrict__ x,
                   const float* __restrict__ Win,
                   float* __restrict__ uproj) {
    __shared__ float  Af32[128 * 32];   // 16 KB, written by TDM
    __shared__ __bf16 Blds[64 * 32];    //  4 KB

    const int tid  = threadIdx.x;
    const int wid  = tid >> 5;
    const int lane = tid & 31;
    const int h    = lane >> 4;   // half-wave: selects K sub-range
    const int l16  = lane & 15;   // row (A) / col (B) within tile
    const int m0   = blockIdx.y * 128;
    const int n0   = blockIdx.x * 64;
    const int wm   = (wid & 3) * 32;   // wave row offset in block tile
    const int wn   = (wid >> 2) * 32;  // wave col offset in block tile

    // LDS byte address of the A staging buffer (low 32 bits of the LDS-
    // aperture generic pointer are the LDS offset).
    const unsigned ldsA = (unsigned)(unsigned long long)(const void*)&Af32[0];

    v8f zero = {};
    v8f acc[2][2];
    acc[0][0] = zero; acc[0][1] = zero; acc[1][0] = zero; acc[1][1] = zero;

    for (int k0 = 0; k0 < NIN; k0 += 32) {
        // ---- A tile: TDM DMA, 2D tile 32(K) x 128(M), f32, one issue/block
        if (wid == 0) {
#if __has_builtin(__builtin_amdgcn_tensor_load_to_lds)
            const unsigned long long ga =
                (unsigned long long)(const void*)(x + (size_t)m0 * NIN + k0);
            v4u g0 = { 1u,                                   // count=1 valid D#
                       ldsA,                                 // lds_addr (bytes)
                       (unsigned)ga,                         // global_addr lo
                       ((unsigned)(ga >> 32) & 0x01FFFFFFu)  // global_addr hi
                           | 0x80000000u };                  // type=2 (image)
            v8i g1 = { (int)(2u << 16),          // data_size = 4 bytes
                       (int)(8192u << 16),       // tensor_dim0[15:0] = NIN
                       (int)(4096u << 16),       // tensor_dim0 hi=0 | tensor_dim1=4096
                       (int)(32u << 16),         // tensor_dim1 hi=0 | tile_dim0=32
                       128,                      // tile_dim1=128, tile_dim2=0
                       8192,                     // tensor_dim0_stride = NIN
                       0, 0 };
            v4i gz4 = { 0, 0, 0, 0 };
            v8i gz8 = { 0, 0, 0, 0, 0, 0, 0, 0 };
            // 6-arg toolchain variant: (g0, g1, g2, g3, g4, cpol)
            __builtin_amdgcn_tensor_load_to_lds(g0, g1, gz4, gz4, gz8, 0);
#else
            // fallback: manual staging by wave 0
            for (int e = lane; e < 128 * 8; e += 32) {
                int row = e >> 3, kq = (e & 7) << 2;
                *(float4*)&Af32[row * 32 + kq] =
                    *(const float4*)(x + (size_t)(m0 + row) * NIN + k0 + kq);
            }
#endif
        }

        // ---- B tile: load f32, convert to bf16, store transposed [n][k]
        if (k0 + 32 < NIN)
            __builtin_prefetch(Win + (size_t)(k0 + 32 + (tid >> 3)) * NOUT
                                   + n0 + (tid & 7) * 8, 0, 1);
        #pragma unroll
        for (int i = 0; i < 8; ++i) {
            int e  = tid + i * 256;       // 2048 elements
            int kk = e >> 6;              // 0..31
            int nn = e & 63;              // 0..63 (consecutive lanes -> coalesced)
            float f = Win[(size_t)(k0 + kk) * NOUT + n0 + nn];
            Blds[nn * 32 + kk] = (__bf16)f;
        }

#if __has_builtin(__builtin_amdgcn_s_wait_tensorcnt)
        __builtin_amdgcn_s_wait_tensorcnt(0);
#endif
        __syncthreads();

        #pragma unroll
        for (int tm = 0; tm < 2; ++tm) {
            const int R = wm + tm * 16 + l16;  // A row for this lane
            // A frag: f32 from LDS -> bf16.  K = h*16..h*16+15
            union { float4 q[4]; float f[16]; } af;
            af.q[0] = *(const float4*)&Af32[R * 32 + h * 16];
            af.q[1] = *(const float4*)&Af32[R * 32 + h * 16 + 4];
            af.q[2] = *(const float4*)&Af32[R * 32 + h * 16 + 8];
            af.q[3] = *(const float4*)&Af32[R * 32 + h * 16 + 12];
            v16bf av;
            #pragma unroll
            for (int i = 0; i < 16; ++i) av[i] = (__bf16)af.f[i];

            #pragma unroll
            for (int tn = 0; tn < 2; ++tn) {
                const int C = wn + tn * 16 + l16;  // B col for this lane
                v16bf b = *(const v16bf*)(&Blds[C * 32 + h * 16]); // K=h*16..+15
                acc[tm][tn] = __builtin_amdgcn_wmma_f32_16x16x32_bf16(
                    false, av, false, b, (short)0, acc[tm][tn], false, false);
            }
        }
        __syncthreads();
    }

    // C/D layout: VGPR r -> M = r (lanes 0-15) / r+8 (lanes 16-31), N = lane%16
    #pragma unroll
    for (int tm = 0; tm < 2; ++tm)
        #pragma unroll
        for (int tn = 0; tn < 2; ++tn)
            #pragma unroll
            for (int r = 0; r < 8; ++r) {
                int M = m0 + wm + tm * 16 + r + h * 8;
                int N = n0 + wn + tn * 16 + l16;
                uproj[(size_t)M * NOUT + N] = acc[tm][tn][r];
            }
}

// ---------------------------------------------------------------------------
// Phase 2: persistent recurrence kernel. 8 blocks x 256 threads = 2048 lanes,
// one output column per lane. W stays L2-resident (16.8 MB < 192 MB).
// One atomic grid barrier per timestep (only 8 arrivals).
// ---------------------------------------------------------------------------
#define P2_BLOCKS  8
#define P2_THREADS 256

__global__ __launch_bounds__(256)
void esn_recurrence(const float* __restrict__ W,
                    float* __restrict__ out,       // in: u_proj rows, out: v rows
                    float* __restrict__ vbuf,      // [NOUT] state, pre-zeroed
                    unsigned* __restrict__ bar) {  // [0]=count, [16]=generation
    __shared__ float vlds[NOUT];
    const int n = blockIdx.x * P2_THREADS + threadIdx.x;   // this lane's column
    const float* wcol = W + n;                             // stride-NOUT column walk
    unsigned* count = bar;
    unsigned* gen   = bar + 16;

    for (int t = 0; t < T_STEPS; ++t) {
        for (int i = threadIdx.x; i < NOUT; i += P2_THREADS)
            vlds[i] = vbuf[i];
        __syncthreads();

        float accum = 0.0f;
        #pragma unroll 8
        for (int k = 0; k < NOUT; ++k)
            accum = __builtin_fmaf(vlds[k], wcol[k * NOUT], accum);

        const float up = out[(size_t)t * NOUT + n];
        const float vn = (1.0f - ALPHA) * vlds[n] + ALPHA * tanhf(accum + up);
        out[(size_t)t * NOUT + n] = vn;
        vbuf[n] = vn;
        __threadfence();
        __syncthreads();

        if (threadIdx.x == 0) {
            unsigned g = __hip_atomic_load(gen, __ATOMIC_ACQUIRE, __HIP_MEMORY_SCOPE_AGENT);
            unsigned arrived = __hip_atomic_fetch_add(count, 1u, __ATOMIC_ACQ_REL,
                                                      __HIP_MEMORY_SCOPE_AGENT) + 1u;
            if (arrived == P2_BLOCKS) {
                __hip_atomic_store(count, 0u, __ATOMIC_RELAXED, __HIP_MEMORY_SCOPE_AGENT);
                __hip_atomic_fetch_add(gen, 1u, __ATOMIC_ACQ_REL, __HIP_MEMORY_SCOPE_AGENT);
            } else {
                while (__hip_atomic_load(gen, __ATOMIC_ACQUIRE, __HIP_MEMORY_SCOPE_AGENT) == g)
                    __builtin_amdgcn_s_sleep(1);
            }
        }
        __syncthreads();
        __threadfence();
    }
}

// ---------------------------------------------------------------------------
extern "C" void kernel_launch(void* const* d_in, const int* in_sizes, int n_in,
                              void* d_out, int out_size, void* d_ws, size_t ws_size,
                              hipStream_t stream) {
    const float* x   = (const float*)d_in[0];   // [T, NIN]
    const float* W   = (const float*)d_in[1];   // [NOUT, NOUT]
    const float* Win = (const float*)d_in[2];   // [NIN, NOUT]
    float* out = (float*)d_out;                 // [T, NOUT]: u_proj then v_t rows

    unsigned* bar  = (unsigned*)d_ws;                   // 256 B barrier region
    float*    vbuf = (float*)((char*)d_ws + 256);       // [NOUT] state

    (void)hipMemsetAsync(d_ws, 0, 256 + NOUT * sizeof(float), stream);

    dim3 g1(NOUT / 64, T_STEPS / 128);   // 32 x 32 blocks
    esn_gemm_bf16<<<g1, dim3(256), 0, stream>>>(x, Win, out);

    esn_recurrence<<<dim3(P2_BLOCKS), dim3(P2_THREADS), 0, stream>>>(W, out, vbuf, bar);
}
